// T5Attention_19765439497099
// MI455X (gfx1250) — matrix operand order repaired and checked by reading, not compiled, not run on hardware
//
#include <hip/hip_runtime.h>
#include <hip/hip_bf16.h>

typedef __attribute__((ext_vector_type(16))) _Float16 v16h;
typedef __attribute__((ext_vector_type(8)))  _Float16 v8h;
typedef __attribute__((ext_vector_type(4)))  _Float16 v4h;
typedef __attribute__((ext_vector_type(2)))  _Float16 v2h;
typedef __attribute__((ext_vector_type(8)))  float    v8f;
typedef __attribute__((ext_vector_type(4)))  float    v4f;
typedef __attribute__((ext_vector_type(4)))  int      v4i;

// ---------------------------------------------------------------------------
// CDNA5 async global->LDS staging (ASYNCcnt path). The builtin takes
// (v4i AS1* gsrc, v4i AS3* ldst, imm offset, imm cpol) per hipcc diagnostics.
// Guarded so the code degrades to the VGPR copy path if builtins are absent.
// ---------------------------------------------------------------------------
#if defined(__has_builtin)
#if __has_builtin(__builtin_amdgcn_global_load_async_to_lds_b128) && \
    __has_builtin(__builtin_amdgcn_s_wait_asynccnt)
#define USE_ASYNC_LDS 1
#endif
#endif
#ifndef USE_ASYNC_LDS
#define USE_ASYNC_LDS 0
#endif

typedef __attribute__((address_space(1))) v4i gas_v4i;
typedef __attribute__((address_space(3))) v4i las_v4i;

static __device__ __forceinline__ void async_cp_b128(const _Float16* g, _Float16* l) {
#if USE_ASYNC_LDS
    __builtin_amdgcn_global_load_async_to_lds_b128((gas_v4i*)g, (las_v4i*)l, 0, 0);
#endif
}
// wait for this wave's outstanding async LDS writes, then block barrier
static __device__ __forceinline__ void publish_barrier() {
#if USE_ASYNC_LDS
    __builtin_amdgcn_s_wait_asynccnt(0);
#endif
    __syncthreads();
}

// ---------------------------------------------------------------------------
// Fragment loaders per CDNA5 ISA 16-bit WMMA layouts (CDNA5 ISA):
//  A 16x32 f16: lane L holds row M=L%16; elems 0..7 -> K=8*(L/16)+0..7,
//               elems 8..15 -> K=16+8*(L/16)+0..7 (two 16B chunks)
//  B 32x16 f16: lane L holds col N=L%16; same K order as A (two 16B chunks)
//  C/D 16x16 f32: vgpr r of lane L = C[M = r + 8*(L/16)][N = L%16]
// ---------------------------------------------------------------------------
static __device__ __forceinline__ v16h load_a_frag(const _Float16* rowbase, int koff, int h) {
    v8h c0 = *(const v8h*)(rowbase + koff + 8 * h);
    v8h c1 = *(const v8h*)(rowbase + koff + 16 + 8 * h);
    v16h t;
#pragma unroll
    for (int e = 0; e < 8; ++e) { t[e] = c0[e]; t[e + 8] = c1[e]; }
    return t;
}
static __device__ __forceinline__ v16h load_b_frag(const _Float16* colbase, int koff, int h) {
    v8h c0 = *(const v8h*)(colbase + koff + 8 * h);
    v8h c1 = *(const v8h*)(colbase + koff + 16 + 8 * h);
    v16h t;
#pragma unroll
    for (int e = 0; e < 8; ++e) { t[e] = c0[e]; t[e + 8] = c1[e]; }
    return t;
}

static __device__ __forceinline__ float rmax16(float v) {
#pragma unroll
    for (int m = 1; m < 16; m <<= 1) v = fmaxf(v, __shfl_xor(v, m, 32));
    return v;
}
static __device__ __forceinline__ float rsum16(float v) {
#pragma unroll
    for (int m = 1; m < 16; m <<= 1) v += __shfl_xor(v, m, 32);
    return v;
}

// ---------------------------------------------------------------------------
// Kernel 1: f32 -> f16 conversion, 8 elem/thread
// ---------------------------------------------------------------------------
__global__ __launch_bounds__(256) void cvt_f32_to_f16(const float* __restrict__ src,
                                                      _Float16* __restrict__ dst, int n) {
    int i = (blockIdx.x * 256 + threadIdx.x) * 8;
    if (i < n) {
        v4f a = *(const v4f*)(src + i);
        v4f b = *(const v4f*)(src + i + 4);
        v8h o;
#pragma unroll
        for (int j = 0; j < 4; ++j) { o[j] = (_Float16)a[j]; o[j + 4] = (_Float16)b[j]; }
        *(v8h*)(dst + i) = o;
    }
}

// ---------------------------------------------------------------------------
// Kernel 2: WMMA GEMM  C[8192,1024] = A[8192,1024] * B[1024,1024]  (f16 in)
//   MODE 0: C -> f16 scattered to [B=8,H=16,S=1024,DK=64]  (q/k/v)
//   MODE 1: C -> f32 row-major (final output)
// 128x128 block tile, 8 waves (4 M x 2 N), wave tile 32x64, K-step 32,
// double-buffered LDS (1 barrier per K-step), async A-tile staging,
// global_prefetch 2 tiles ahead.
// ---------------------------------------------------------------------------
#define GT_M 128
#define GT_N 128
#define GT_K 32
#define GN   1024
#define GK   1024
#define LDS_STR 40  // padded f16 stride (80B)

static __device__ __forceinline__ void stage_tiles(const _Float16* __restrict__ A,
                                                   const _Float16* __restrict__ Bw,
                                                   _Float16* sA, _Float16* sB,
                                                   int m0, int n0, int k0, int tid) {
    // A tile: row r, one 16-elem (32B) segment per thread -> row-major LDS
    {
        int r = tid >> 1, sg = (tid & 1) * 16;
        const _Float16* gp = A + (size_t)(m0 + r) * GK + k0 + sg;
        _Float16* lp = &sA[r * LDS_STR + sg];
#if USE_ASYNC_LDS
        async_cp_b128(gp, lp);
        async_cp_b128(gp + 8, lp + 8);
#else
        v8h u0 = *(const v8h*)gp;
        v8h u1 = *(const v8h*)(gp + 8);
        *(v8h*)lp       = u0;
        *(v8h*)(lp + 8) = u1;
#endif
    }
    // B tile transposed [n][k]: k-row pair (2kp,2kp+1) x 8 n per thread,
    // packing (k,k+1) pairs into b32 LDS stores (transpose => VGPR path).
    {
        int kp = tid >> 4;
        int ns = (tid & 15) * 8;
        const _Float16* g0 = Bw + (size_t)(k0 + 2 * kp) * GN + n0 + ns;
        v8h u0 = *(const v8h*)g0;
        v8h u1 = *(const v8h*)(g0 + GN);
#pragma unroll
        for (int j = 0; j < 8; ++j) {
            v2h p; p[0] = u0[j]; p[1] = u1[j];
            *(v2h*)&sB[(ns + j) * LDS_STR + 2 * kp] = p;
        }
    }
}

template <int MODE>
__global__ __launch_bounds__(256) void gemm_wmma_f16(const _Float16* __restrict__ A,
                                                     const _Float16* __restrict__ Bw,
                                                     void* __restrict__ Cout) {
    __shared__ _Float16 sA[2][GT_M * LDS_STR];
    __shared__ _Float16 sB[2][GT_N * LDS_STR];

    const int tid  = threadIdx.x;
    const int lane = tid & 31;
    const int wave = tid >> 5;
    const int wm   = wave & 3;
    const int wn   = wave >> 2;
    const int h    = lane >> 4;
    const int l16  = lane & 15;
    const int m0   = blockIdx.x * GT_M;
    const int n0   = blockIdx.y * GT_N;

    stage_tiles(A, Bw, sA[0], sB[0], m0, n0, 0, tid);
    publish_barrier();

    v8f acc[2][4] = {};
    int buf = 0;
    for (int k0 = 0; k0 < GK; k0 += GT_K, buf ^= 1) {
        if (k0 + GT_K < GK)
            stage_tiles(A, Bw, sA[buf ^ 1], sB[buf ^ 1], m0, n0, k0 + GT_K, tid);
        if (k0 + 2 * GT_K < GK) {
            __builtin_prefetch(A + (size_t)(m0 + (tid >> 1)) * GK + k0 + 2 * GT_K, 0, 1);
            __builtin_prefetch(Bw + (size_t)(k0 + 2 * GT_K + 2 * (tid >> 4)) * GN + n0, 0, 1);
        }

        v16h afrag[2];
#pragma unroll
        for (int i = 0; i < 2; ++i)
            afrag[i] = load_a_frag(&sA[buf][(wm * 32 + i * 16 + l16) * LDS_STR], 0, h);
#pragma unroll
        for (int j = 0; j < 4; ++j) {
            v16h bfrag = load_b_frag(&sB[buf][(wn * 64 + j * 16 + l16) * LDS_STR], 0, h);
#pragma unroll
            for (int i = 0; i < 2; ++i)
                acc[i][j] = __builtin_amdgcn_wmma_f32_16x16x32_f16(
                    false, afrag[i], false, bfrag, (short)0, acc[i][j], false, false);
        }
        publish_barrier();   // async writes to buf^1 complete + readers done
    }

    // ---- epilogue: single base address, constant immediate offsets ----
    if (MODE == 0) {
        int bb = m0 >> 10;
        int hd = (n0 + wn * 64) >> 6;
        int sb = (m0 & 1023) + wm * 32 + 8 * h;
        _Float16* base = (_Float16*)Cout + ((size_t)(bb * 16 + hd) * 1024 + sb) * 64 + l16;
#pragma unroll
        for (int i = 0; i < 2; ++i)
#pragma unroll
            for (int r = 0; r < 8; ++r)
#pragma unroll
                for (int j = 0; j < 4; ++j)
                    base[(i * 16 + r) * 64 + j * 16] = (_Float16)acc[i][j][r];
    } else {
        float* base = (float*)Cout + (size_t)(m0 + wm * 32 + 8 * h) * GN + n0 + wn * 64 + l16;
#pragma unroll
        for (int i = 0; i < 2; ++i)
#pragma unroll
            for (int r = 0; r < 8; ++r)
#pragma unroll
                for (int j = 0; j < 4; ++j)
                    base[(i * 16 + r) * GN + j * 16] = acc[i][j][r];
    }
}

// ---------------------------------------------------------------------------
// Kernel 3: fused flash attention with T5 relative position bias.
// Grid: B*H*(S/64); block = 128 threads = 4 waves; wave owns 16 q-rows.
// ---------------------------------------------------------------------------
#define AT_STR 72  // padded f16 stride (144B)

__device__ __forceinline__ int t5_bucket(int delta /* = k - q */) {
    int n = -delta;
    int ret = 0;
    if (n < 0) { ret = 16; n = -n; }
    if (n < 8) return ret + n;
    float v = __logf((float)n * 0.125f) * (8.0f / __logf(16.0f));
    int vi = 8 + (int)v;
    if (vi > 15) vi = 15;
    return ret + vi;
}

__global__ __launch_bounds__(128) void attn_flash_wmma(const _Float16* __restrict__ Q,
                                                       const _Float16* __restrict__ Km,
                                                       const _Float16* __restrict__ Vm,
                                                       const float* __restrict__ rel_emb,
                                                       _Float16* __restrict__ Ctx) {
    __shared__ _Float16 sQ[64 * AT_STR];     // [q][d]
    __shared__ _Float16 sK[64 * AT_STR];     // [kv][d] (== B^T layout for Q*K^T)
    __shared__ _Float16 sVt[64 * AT_STR];    // [d][kv] (transposed V)
    __shared__ _Float16 sP[4 * 16 * AT_STR]; // per-wave P staging [16][kv]
    __shared__ float    sBias[2048];         // bias[(k-q)+1023]

    const int bid  = blockIdx.x;
    const int qt   = bid & 15;
    const int bh   = bid >> 4;
    const int head = bh & 15;
    const int b    = bh >> 4;
    const int q0   = qt * 64;

    const int tid  = threadIdx.x;
    const int lane = tid & 31;
    const int wave = tid >> 5;
    const int h2   = lane >> 4;
    const int l16  = lane & 15;

    const size_t headoff = (size_t)(b * 16 + head) * 1024 * 64;
    const _Float16* Qb = Q  + headoff;
    const _Float16* Kb = Km + headoff;
    const _Float16* Vb = Vm + headoff;

    for (int idx = tid; idx < 2047; idx += 128)
        sBias[idx] = rel_emb[t5_bucket(idx - 1023) * 16 + head];

    {   // Q tile (64x64): async straight copy
        int r = tid >> 1, sg = (tid & 1) * 32;
        const _Float16* gp = Qb + (size_t)(q0 + r) * 64 + sg;
        _Float16* lp = &sQ[r * AT_STR + sg];
#if USE_ASYNC_LDS
#pragma unroll
        for (int c = 0; c < 4; ++c) async_cp_b128(gp + c * 8, lp + c * 8);
#else
#pragma unroll
        for (int c = 0; c < 4; ++c) ((v8h*)lp)[c] = ((const v8h*)gp)[c];
#endif
    }

    v8f   o[4] = {};
    float mrow[8], lrow[8];
#pragma unroll
    for (int r = 0; r < 8; ++r) { mrow[r] = -1e30f; lrow[r] = 0.0f; }

    for (int kv0 = 0; kv0 < 1024; kv0 += 64) {
        __syncthreads();   // previous iteration done reading sK/sVt
        {   // K tile row-major: async straight copy
            int r = tid >> 1, sg = (tid & 1) * 32;
            const _Float16* gp = Kb + (size_t)(kv0 + r) * 64 + sg;
            _Float16* lp = &sK[r * AT_STR + sg];
#if USE_ASYNC_LDS
#pragma unroll
            for (int c = 0; c < 4; ++c) async_cp_b128(gp + c * 8, lp + c * 8);
#else
#pragma unroll
            for (int c = 0; c < 4; ++c) ((v8h*)lp)[c] = ((const v8h*)gp)[c];
#endif
        }
        {   // V tile transposed with b32 pair-packed stores (VGPR path)
            int rp  = tid >> 2;            // kv rows 2rp, 2rp+1
            int ds0 = (tid & 3) * 16;
            const _Float16* g0 = Vb + (size_t)(kv0 + 2 * rp) * 64 + ds0;
            v8h u0a = ((const v8h*)g0)[0],        u0b = ((const v8h*)g0)[1];
            v8h u1a = ((const v8h*)(g0 + 64))[0], u1b = ((const v8h*)(g0 + 64))[1];
#pragma unroll
            for (int j = 0; j < 8; ++j) {
                v2h p; p[0] = u0a[j]; p[1] = u1a[j];
                *(v2h*)&sVt[(ds0 + j) * AT_STR + 2 * rp] = p;
            }
#pragma unroll
            for (int j = 0; j < 8; ++j) {
                v2h p; p[0] = u0b[j]; p[1] = u1b[j];
                *(v2h*)&sVt[(ds0 + 8 + j) * AT_STR + 2 * rp] = p;
            }
        }
        if (kv0 + 64 < 1024) {
            __builtin_prefetch(Kb + (size_t)(kv0 + 64 + (tid >> 1)) * 64 + (tid & 1) * 32, 0, 1);
            __builtin_prefetch(Vb + (size_t)(kv0 + 64 + 2 * (tid >> 2)) * 64 + (tid & 3) * 16, 0, 1);
        }
        publish_barrier();

        // ---- S = Q * K^T ----
        v8f sc[4] = {};
        v16h aq[2];
#pragma unroll
        for (int ks = 0; ks < 2; ++ks)
            aq[ks] = load_a_frag(&sQ[(wave * 16 + l16) * AT_STR], ks * 32, h2);
#pragma unroll
        for (int j = 0; j < 4; ++j) {
#pragma unroll
            for (int ks = 0; ks < 2; ++ks) {
                v16h bk = load_b_frag(&sK[(j * 16 + l16) * AT_STR], ks * 32, h2);
                sc[j] = __builtin_amdgcn_wmma_f32_16x16x32_f16(
                    false, aq[ks], false, bk, (short)0, sc[j], false, false);
            }
        }

        // ---- bias + online softmax ----
#pragma unroll
        for (int r = 0; r < 8; ++r) {
            int m = q0 + wave * 16 + r + 8 * h2;
            int bidx = kv0 - m + 1023 + l16;
            float rowmax = -1e30f;
#pragma unroll
            for (int j = 0; j < 4; ++j) {
                float s = sc[j][r] + sBias[bidx + j * 16];
                sc[j][r] = s;
                rowmax = fmaxf(rowmax, s);
            }
            rowmax = rmax16(rowmax);
            float newm  = fmaxf(mrow[r], rowmax);
            float alpha = __expf(mrow[r] - newm);
            float rs = 0.0f;
#pragma unroll
            for (int j = 0; j < 4; ++j) {
                float p = __expf(sc[j][r] - newm);
                sc[j][r] = p;
                rs += p;
            }
            rs = rsum16(rs);
            lrow[r] = lrow[r] * alpha + rs;
            mrow[r] = newm;
#pragma unroll
            for (int j = 0; j < 4; ++j) o[j][r] *= alpha;
        }

        // ---- stage P (C-layout -> A-layout); wave-private region: in-order
        //      LDS ops + compiler s_wait_dscnt make this barrier-free. ----
        _Float16* pw = &sP[wave * 16 * AT_STR];
#pragma unroll
        for (int j = 0; j < 4; ++j)
#pragma unroll
            for (int r = 0; r < 8; ++r)
                pw[(r + 8 * h2) * AT_STR + j * 16 + l16] = (_Float16)sc[j][r];

        // ---- O += P * V ----
#pragma unroll
        for (int ks = 0; ks < 2; ++ks) {
            v16h ap = load_a_frag(&pw[l16 * AT_STR], ks * 32, h2);
#pragma unroll
            for (int j = 0; j < 4; ++j) {
                v16h bv = load_b_frag(&sVt[(j * 16 + l16) * AT_STR], ks * 32, h2);
                o[j] = __builtin_amdgcn_wmma_f32_16x16x32_f16(
                    false, ap, false, bv, (short)0, o[j], false, false);
            }
        }
    }

    // ---- normalize, store ctx f16 [B,S,H*DK]; one base + immediate offsets ----
    _Float16* base = Ctx + ((size_t)(b * 1024) + q0 + wave * 16 + 8 * h2) * 1024
                         + head * 64 + l16;
#pragma unroll
    for (int r = 0; r < 8; ++r) {
        float inv = 1.0f / lrow[r];
#pragma unroll
        for (int j = 0; j < 4; ++j)
            base[r * 1024 + j * 16] = (_Float16)(o[j][r] * inv);
    }
}

// ---------------------------------------------------------------------------
extern "C" void kernel_launch(void* const* d_in, const int* in_sizes, int n_in,
                              void* d_out, int out_size, void* d_ws, size_t ws_size,
                              hipStream_t stream) {
    const float* x   = (const float*)d_in[0];
    const float* Wq  = (const float*)d_in[1];
    const float* Wk  = (const float*)d_in[2];
    const float* Wv  = (const float*)d_in[3];
    const float* Wo  = (const float*)d_in[4];
    const float* rel = (const float*)d_in[5];

    char* ws = (char*)d_ws;
    const size_t MB = 1u << 20;
    _Float16* xb  = (_Float16*)(ws);
    _Float16* wqh = (_Float16*)(ws + 16 * MB);
    _Float16* wkh = (_Float16*)(ws + 18 * MB);
    _Float16* wvh = (_Float16*)(ws + 20 * MB);
    _Float16* woh = (_Float16*)(ws + 22 * MB);
    _Float16* qh  = (_Float16*)(ws + 24 * MB);
    _Float16* kh  = (_Float16*)(ws + 40 * MB);
    _Float16* vh  = (_Float16*)(ws + 56 * MB);
    _Float16* ctx = (_Float16*)(ws + 72 * MB);

    const int NX = 8 * 1024 * 1024;
    const int NW = 1024 * 1024;
    cvt_f32_to_f16<<<NX / 8 / 256, 256, 0, stream>>>(x, xb, NX);
    cvt_f32_to_f16<<<NW / 8 / 256, 256, 0, stream>>>(Wq, wqh, NW);
    cvt_f32_to_f16<<<NW / 8 / 256, 256, 0, stream>>>(Wk, wkh, NW);
    cvt_f32_to_f16<<<NW / 8 / 256, 256, 0, stream>>>(Wv, wvh, NW);
    cvt_f32_to_f16<<<NW / 8 / 256, 256, 0, stream>>>(Wo, woh, NW);

    dim3 gg(8192 / GT_M, 1024 / GT_N);
    gemm_wmma_f16<0><<<gg, 256, 0, stream>>>(xb, wqh, qh);
    gemm_wmma_f16<0><<<gg, 256, 0, stream>>>(xb, wkh, kh);
    gemm_wmma_f16<0><<<gg, 256, 0, stream>>>(xb, wvh, vh);

    attn_flash_wmma<<<8 * 16 * 16, 128, 0, stream>>>(qh, kh, vh, rel, ctx);

    gemm_wmma_f16<1><<<gg, 256, 0, stream>>>(ctx, woh, d_out);
}
